// DorefaConv2d_6279242186832
// MI455X (gfx1250) — 8-bit matrix operand order repaired and checked by reading, compiled, not run on hardware
//
#include <hip/hip_runtime.h>
#include <hip/hip_bf16.h>
#include <stdint.h>


typedef int v8i __attribute__((ext_vector_type(8)));

union V8 {
    v8i      v;
    int      e[8];
    uint32_t u[8];
};

__global__ __launch_bounds__(256) void dorefa_meanabs(const float* __restrict__ W,
                                                      float* __restrict__ e_out) {
    __shared__ float red[256];
    float s = 0.0f;
    for (int i = threadIdx.x; i < 36864; i += 256) s += fabsf(W[i]);
    red[threadIdx.x] = s;
    __syncthreads();
    for (int off = 128; off > 0; off >>= 1) {
        if ((int)threadIdx.x < off) red[threadIdx.x] += red[threadIdx.x + off];
        __syncthreads();
    }
    if (threadIdx.x == 0) e_out[0] = red[0] * (1.0f / 36864.0f);
}

__global__ __launch_bounds__(256) void dorefa_packB(const float* __restrict__ W,
                                                    uint32_t* __restrict__ bpack) {
    int id = blockIdx.x * 256 + threadIdx.x;
    if (id >= 9216) return;
    int v    = id & 7;
    int lane = (id >> 3) & 31;
    int j    = (id >> 8) & 3;
    int t    = id >> 10;
    int nn   = lane & 15;
    int half = lane >> 4;
    int kbase = ((v >> 2) & 1) * 32 + half * 16 + (v & 3) * 4;
    uint32_t word = 0;
#pragma unroll
    for (int b = 0; b < 4; ++b) {
        int cin = kbase + b;
        float w = W[((size_t)t * 64 + cin) * 64 + (j * 16 + nn)];
        int s = (w > 0.0f) ? 1 : ((w < 0.0f) ? -1 : 0);
        word |= ((uint32_t)(uint8_t)(int8_t)s) << (8 * b);
    }
    bpack[id] = word;
}

__global__ __launch_bounds__(256) void dorefa_conv(const float* __restrict__ x,
                                                   const float* __restrict__ bias,
                                                   const uint32_t* __restrict__ bpack,
                                                   const float* __restrict__ e_ws,
                                                   float* __restrict__ out) {
    __shared__ __align__(16) uint32_t s_act[3 * 130 * 16];
    __shared__ __align__(16) uint32_t s_b[9216];

    const int tid = threadIdx.x;
    const int blk = blockIdx.x;
    const int n = blk >> 7;
    const int h = blk & 127;

    {
        const uint4* bsrc = (const uint4*)bpack;
        uint4* bdst = (uint4*)s_b;
        for (int i = tid; i < 2304; i += 256) bdst[i] = bsrc[i];
    }

    for (int i = tid; i < 6240; i += 256) {
        int pi = i >> 4;
        int c4 = i & 15;
        int r = pi / 130;
        int c = (pi % 130) - 1;
        int row = h + r - 1;
        uint32_t packed = 0;
        if (row >= 0 && row < 128 && c >= 0 && c < 128) {
            const float* src = x + ((((size_t)n * 128 + row) * 128 + c) * 64 + c4 * 4);
#pragma unroll
            for (int b = 0; b < 4; ++b) {
                float v = fminf(1.0f, fabsf(src[b]));
                uint32_t q = (uint32_t)__builtin_rintf(v * 7.0f);
                packed |= (q & 0xFFu) << (8 * b);
            }
        }
        s_act[i] = packed;
    }
    __syncthreads();

    const int wave = tid >> 5;
    const int lane = tid & 31;
    const int n16  = lane & 15;
    const int half = lane >> 4;

    V8 acc[4];
#pragma unroll
    for (int j = 0; j < 4; ++j)
#pragma unroll
        for (int r = 0; r < 8; ++r) acc[j].e[r] = 0;

#pragma unroll
    for (int dy = 0; dy < 3; ++dy) {
#pragma unroll
        for (int dx = 0; dx < 3; ++dx) {
            const int t = dy * 3 + dx;
            const int pix = dy * 130 + wave * 16 + n16 + dx;
            const uint32_t* ap = &s_act[pix * 16 + half * 4];
            V8 a;
            a.u[0] = ap[0];  a.u[1] = ap[1];
            a.u[2] = ap[2];  a.u[3] = ap[3];
            a.u[4] = ap[8];  a.u[5] = ap[9];
            a.u[6] = ap[10]; a.u[7] = ap[11];
#pragma unroll
            for (int j = 0; j < 4; ++j) {
                const uint32_t* bp = &s_b[((t * 4 + j) * 32 + lane) * 8];
                V8 bf;
#pragma unroll
                for (int v = 0; v < 8; ++v) bf.u[v] = bp[v];
                acc[j].v = __builtin_amdgcn_wmma_i32_16x16x64_iu8(
                     false, a.v,  true, bf.v, acc[j].v,
                     false,  false);
            }
        }
    }

    const float scale = e_ws[0] * (1.0f / 7.0f);
    const size_t rowbase = ((size_t)n * 128 + h) * 128;
#pragma unroll
    for (int j = 0; j < 4; ++j) {
        const float bv = bias[j * 16 + n16];
#pragma unroll
        for (int r = 0; r < 8; ++r) {
            const int M = r + half * 8;
            const size_t p = rowbase + (size_t)(wave * 16 + M);
            out[p * 64 + j * 16 + n16] = (float)acc[j].e[r] * scale + bv;
        }
    }
}

extern "C" void kernel_launch(void* const* d_in, const int* in_sizes, int n_in,
                              void* d_out, int out_size, void* d_ws, size_t ws_size,
                              hipStream_t stream) {
    const float* x = (const float*)d_in[0];
    const float* W = (const float*)d_in[1];
    const float* b = (const float*)d_in[2];
    float* out = (float*)d_out;

    float* e_ws = (float*)d_ws;
    uint32_t* bpack = (uint32_t*)((char*)d_ws + 256);

    dorefa_meanabs<<<1, 256, 0, stream>>>(W, e_ws);
    dorefa_packB<<<36, 256, 0, stream>>>(W, bpack);
    dorefa_conv<<<4096, 256, 0, stream>>>(x, b, bpack, e_ws, out);
}
